// twoStream_multiView_Net_24644522344717
// MI455X (gfx1250) — matrix operand order repaired and checked by reading, not compiled, not run on hardware
//
#include <hip/hip_runtime.h>
#include <hip/hip_bf16.h>
#include <math.h>

// ---------------------------------------------------------------------------
// CDNA5 WMMA (wave32): D(16x16 f32) = A(16x32 bf16) x B(32x16 bf16) + C
// Activations are NHWC bf16; implicit-GEMM k-order is (r,s,c) so that 8
// consecutive k == 8 consecutive channels == one 16-byte global load.
// ---------------------------------------------------------------------------
typedef __attribute__((ext_vector_type(16))) __bf16       v16bf;
typedef __attribute__((ext_vector_type(8)))  float        v8f;
typedef __attribute__((ext_vector_type(4)))  unsigned int v4u;

#define BN_EPS 1e-5f

__device__ __forceinline__ float  bf2f(__bf16 b) { return (float)b; }
__device__ __forceinline__ __bf16 f2bf(float f)  { return (__bf16)f; }

// ---------------------------------------------------------------------------
// Preprocessing
// ---------------------------------------------------------------------------
__global__ void view_mean_kernel(const float* __restrict__ x, float* __restrict__ m, int CHW) {
    __shared__ float red[256];
    const float* p = x + (size_t)blockIdx.x * CHW;
    float s = 0.f;
    for (int i = threadIdx.x; i < CHW; i += 256) s += p[i];
    red[threadIdx.x] = s;
    __syncthreads();
    for (int off = 128; off > 0; off >>= 1) {
        if (threadIdx.x < off) red[threadIdx.x] += red[threadIdx.x + off];
        __syncthreads();
    }
    if (threadIdx.x == 0) m[blockIdx.x] = red[0] / (float)CHW;
}

// rank[v] = #{j : m[v] < m[j]}; perm[rank] = v (inverse permutation)
__global__ void rank_perm_kernel(const float* __restrict__ m, int* __restrict__ perm, int B, int V) {
    int t = threadIdx.x;
    if (t >= B * V) return;
    int b = t / V, v = t % V;
    float mv = m[b * V + v];
    int rank = 0;
    for (int j = 0; j < V; ++j) rank += (mv < m[b * V + j]) ? 1 : 0;
    perm[b * V + rank] = v;
}

// x: NCHW f32 -> xs, xr: NHWC bf16 (sorted views / |avg - view|)
__global__ void sort_avg_diff_kernel(const float* __restrict__ x, const int* __restrict__ perm,
                                     __bf16* __restrict__ xs, __bf16* __restrict__ xr,
                                     int B, int V, int C, int HWpix) {
    int idx = blockIdx.x * blockDim.x + threadIdx.x;
    if (idx >= B * HWpix) return;
    int b  = idx / HWpix;
    int hw = idx % HWpix;
    float vals[8][3];
    float avg[3] = {0.f, 0.f, 0.f};
    #pragma unroll
    for (int v = 0; v < 8; ++v) {
        int src = perm[b * V + v];
        #pragma unroll
        for (int c = 0; c < 3; ++c) {
            float val = x[(((size_t)(b * V + src) * C + c) * HWpix) + hw];
            vals[v][c] = val;
            avg[c] += val;
        }
    }
    #pragma unroll
    for (int c = 0; c < 3; ++c) avg[c] *= (1.f / 8.f);
    #pragma unroll
    for (int v = 0; v < 8; ++v) {
        size_t o = ((size_t)(b * V + v) * HWpix + hw) * C;
        #pragma unroll
        for (int c = 0; c < 3; ++c) {
            xs[o + c] = f2bf(vals[v][c]);
            xr[o + c] = f2bf(fabsf(avg[c] - vals[v][c]));
        }
    }
}

// ---------------------------------------------------------------------------
// Weight packing: W (OIHW f32, [K][C][R][S]) -> B-fragment stream (bf16),
// k-order (r,s,c): k = (r*S + s)*C + c, zero-padded to nKS*32.
//   fragment (nt, ks): 32 lanes x 16 bf16; col n = lane&15, K-half = lane>>4
// ---------------------------------------------------------------------------
__global__ void pack_b_kernel(const float* __restrict__ W, __bf16* __restrict__ Bp,
                              int K, int C, int R, int S, int nKS) {
    int t = blockIdx.x * blockDim.x + threadIdx.x;
    int total = (K >> 4) * nKS * 32;
    if (t >= total) return;
    int lane = t & 31;
    int ks   = (t >> 5) % nKS;
    int nt   = t / (32 * nKS);
    int n    = (nt << 4) + (lane & 15);
    int half = lane >> 4;
    int KK   = C * R * S;
    v16bf out;
    #pragma unroll
    for (int pos = 0; pos < 16; ++pos) {
        int k = ks * 32 + half * 8 + (pos & 7) + ((pos >> 3) << 4);
        float v = 0.f;
        if (k < KK) {
            int c  = k % C;
            int rs = k / C;
            int s  = rs % S;
            int r  = rs / S;
            v = W[(((size_t)n * C + c) * R + r) * S + s];
        }
        out[pos] = f2bf(v);
    }
    ((v16bf*)Bp)[(size_t)t] = out;   // 32B contiguous store
}

// ---------------------------------------------------------------------------
// Implicit-GEMM conv (NHWC), fused BN (+residual)(+ReLU)
//   M = N*P*Q, Ndim = K (out ch), Kdim = R*S*C (padded to nKS*32)
//   block = 128 thr = 4 waves; wave tile = 32(M) x 64(N) -> 8 WMMA per K-step
//   A: fast path (C%8==0): one global_load_b128 per 16B fragment slot,
//      chunk (c0,r,s) state carried incrementally across K-steps (no divides
//      in hot loop); staged into ping-pong LDS (ds b128), pipelined 1 ahead.
//   B: packed global fragments, strided pointer bumps + global_prefetch_b8.
// ---------------------------------------------------------------------------
struct ConvParams {
    const __bf16* X;    // [N, H, W, C]
    const __bf16* Bp;   // packed weight fragments
    const float*  bn_g;
    const float*  bn_b;
    const float*  bn_m;
    const float*  bn_v;
    const __bf16* res;  // residual [N, P, Q, K] or null
    __bf16*       Y;    // [N, P, Q, K]
    int N, C, H, W, K, R, S, stride, pad, P, Q, relu, nKS;
};

__launch_bounds__(128)
__global__ void conv_wmma_kernel(ConvParams p) {
    const int lane = threadIdx.x & 31;
    const int wave = threadIdx.x >> 5;            // 0..3 -> M sub-range
    const int tile_m0 = blockIdx.x * 128;
    const int tile_n0 = blockIdx.y * 64;
    const int M  = p.N * p.P * p.Q;
    const int KK = p.R * p.S * p.C;
    const bool fastC = ((p.C & 7) == 0) && (p.C >= 32);

    // ping-pong fragment-order A staging: 2 x 8 M-subtiles x 32 lanes x 32B
    __shared__ v16bf AsFrag[2][8][32];
    v4u* AsV = (v4u*)AsFrag;                       // 512 v4u per buffer

    // ---- K-invariant per-thread staging state (4 chunks of 8 k-elements) ----
    int s_dst[4], s_kg[4], s_nH[4], s_hb[4], s_wb[4], s_valid[4];
    int f_c[4], f_r[4], f_s[4];                    // fast-path incremental state
    #pragma unroll
    for (int ch = 0; ch < 4; ++ch) {
        int i      = threadIdx.x + (ch << 7);
        int mlocal = i >> 2;                       // 0..127
        int kg     = (i & 3) << 3;                 // 0,8,16,24
        int sub      = mlocal >> 4;
        int lane_dst = (mlocal & 15) + ((kg & 8) << 1);
        s_dst[ch] = (sub * 32 + lane_dst) * 2 + ((kg >> 4) & 1);
        s_kg[ch]  = kg;
        f_c[ch] = kg; f_r[ch] = 0; f_s[ch] = 0;    // kg < 32 <= C on fast path
        int m = tile_m0 + mlocal;
        if (m < M) {
            int q  = m % p.Q; int t = m / p.Q;
            int pp = t % p.P; int n = t / p.P;
            s_valid[ch] = 1;
            s_nH[ch]    = n * p.H;
            s_hb[ch]    = pp * p.stride - p.pad;
            s_wb[ch]    = q  * p.stride - p.pad;
        } else {
            s_valid[ch] = 0; s_nH[ch] = 0; s_hb[ch] = 0; s_wb[ch] = 0;
        }
    }

    // stage K-step ks into buffer bufsel; must be called for ks = 0,1,2,...
    // exactly once each (fast-path state advances by 32 k per call).
    auto stage = [&](int ks, int bufsel) {
        const int k0 = ks * 32;
        if (fastC) {
            #pragma unroll
            for (int ch = 0; ch < 4; ++ch) {
                v4u pk = {0u, 0u, 0u, 0u};
                if (s_valid[ch] && (k0 + s_kg[ch]) < KK) {
                    int hin = s_hb[ch] + f_r[ch];
                    int win = s_wb[ch] + f_s[ch];
                    if (hin >= 0 && hin < p.H && win >= 0 && win < p.W)
                        pk = *(const v4u*)(p.X +
                             ((size_t)(s_nH[ch] + hin) * p.W + win) * p.C + f_c[ch]);
                }
                AsV[bufsel * 512 + s_dst[ch]] = pk;        // ds_store_b128
                // advance chunk k by 32 (32 <= C, so at most one (r,s) carry)
                f_c[ch] += 32;
                if (f_c[ch] >= p.C) {
                    f_c[ch] -= p.C;
                    ++f_s[ch];
                    if (f_s[ch] == p.S) { f_s[ch] = 0; ++f_r[ch]; }
                }
            }
        } else {   // generic path (conv1: C=3), per-element gather
            #pragma unroll
            for (int ch = 0; ch < 4; ++ch) {
                int kk = k0 + s_kg[ch];
                int c  = kk % p.C; int rs = kk / p.C;
                int s  = rs % p.S; int r  = rs / p.S;
                unsigned short us[8];
                #pragma unroll
                for (int tt = 0; tt < 8; ++tt) {
                    unsigned short uv = 0;
                    if (s_valid[ch] && (kk + tt) < KK) {
                        int hin = s_hb[ch] + r;
                        int win = s_wb[ch] + s;
                        if (hin >= 0 && hin < p.H && win >= 0 && win < p.W)
                            uv = __builtin_bit_cast(unsigned short,
                                 p.X[((size_t)(s_nH[ch] + hin) * p.W + win) * p.C + c]);
                    }
                    us[tt] = uv;
                    ++c;
                    if (c == p.C) { c = 0; ++s; if (s == p.S) { s = 0; ++r; } }
                }
                v4u pk;
                pk[0] = (unsigned)us[0] | ((unsigned)us[1] << 16);
                pk[1] = (unsigned)us[2] | ((unsigned)us[3] << 16);
                pk[2] = (unsigned)us[4] | ((unsigned)us[5] << 16);
                pk[3] = (unsigned)us[6] | ((unsigned)us[7] << 16);
                AsV[bufsel * 512 + s_dst[ch]] = pk;
            }
        }
    };

    // ---- B fragment pointers (one per 16-wide N subtile) ----
    const v16bf* __restrict__ Bv = (const v16bf*)p.Bp;
    const size_t bstride = (size_t)p.nKS * 32;
    const v16bf* bp0 = Bv + (size_t)(tile_n0 >> 4) * bstride + lane;
    const v16bf* bp1 = bp0 + bstride;
    const v16bf* bp2 = bp1 + bstride;
    const v16bf* bp3 = bp2 + bstride;

    v8f acc[2][4] = {};

    stage(0, 0);
    __syncthreads();

    for (int ks = 0; ks < p.nKS; ++ks) {
        const int cur = ks & 1;
        v16bf a0 = AsFrag[cur][wave * 2 + 0][lane];     // ds_load_b128 x2
        v16bf a1 = AsFrag[cur][wave * 2 + 1][lane];
        if (ks + 1 < p.nKS) stage(ks + 1, cur ^ 1);     // pipeline next step

        v16bf b0 = *bp0, b1 = *bp1, b2 = *bp2, b3 = *bp3;
        acc[0][0] = __builtin_amdgcn_wmma_f32_16x16x32_bf16(false, a0, false, b0, (short)0, acc[0][0], false, false);
        acc[1][0] = __builtin_amdgcn_wmma_f32_16x16x32_bf16(false, a1, false, b0, (short)0, acc[1][0], false, false);
        acc[0][1] = __builtin_amdgcn_wmma_f32_16x16x32_bf16(false, a0, false, b1, (short)0, acc[0][1], false, false);
        acc[1][1] = __builtin_amdgcn_wmma_f32_16x16x32_bf16(false, a1, false, b1, (short)0, acc[1][1], false, false);
        acc[0][2] = __builtin_amdgcn_wmma_f32_16x16x32_bf16(false, a0, false, b2, (short)0, acc[0][2], false, false);
        acc[1][2] = __builtin_amdgcn_wmma_f32_16x16x32_bf16(false, a1, false, b2, (short)0, acc[1][2], false, false);
        acc[0][3] = __builtin_amdgcn_wmma_f32_16x16x32_bf16(false, a0, false, b3, (short)0, acc[0][3], false, false);
        acc[1][3] = __builtin_amdgcn_wmma_f32_16x16x32_bf16(false, a1, false, b3, (short)0, acc[1][3], false, false);

        bp0 += 32; bp1 += 32; bp2 += 32; bp3 += 32;
        __builtin_prefetch(bp0, 0, 1);                  // speculative prefetch
        __syncthreads();
    }

    // ---- epilogue: BN + (residual) + (ReLU), NHWC output ----
    // D layout: lane holds col n = lane&15; rows m16 = 8*(lane>>4) + r
    float scl[4], shf[4];
    #pragma unroll
    for (int ni = 0; ni < 4; ++ni) {
        int nch = tile_n0 + ni * 16 + (lane & 15);
        if (nch < p.K) {
            scl[ni] = p.bn_g[nch] * rsqrtf(p.bn_v[nch] + BN_EPS);
            shf[ni] = p.bn_b[nch] - p.bn_m[nch] * scl[ni];
        } else { scl[ni] = 0.f; shf[ni] = 0.f; }
    }
    #pragma unroll
    for (int mi = 0; mi < 2; ++mi) {
        int mbase = tile_m0 + wave * 32 + mi * 16 + (lane >> 4) * 8;
        int q  = mbase % p.Q; int t = mbase / p.Q;
        int pp = t % p.P;     int n = t / p.P;
        #pragma unroll
        for (int r = 0; r < 8; ++r) {
            if (mbase + r < M) {
                size_t spatial = ((size_t)(n * p.P + pp) * p.Q + q) * p.K;
                #pragma unroll
                for (int ni = 0; ni < 4; ++ni) {
                    int nch = tile_n0 + ni * 16 + (lane & 15);
                    if (nch < p.K) {
                        size_t o = spatial + nch;
                        float v = acc[mi][ni][r] * scl[ni] + shf[ni];
                        if (p.res)  v += bf2f(p.res[o]);
                        if (p.relu) v  = fmaxf(v, 0.f);
                        p.Y[o] = f2bf(v);
                    }
                }
            }
            ++q;
            if (q == p.Q) { q = 0; ++pp; if (pp == p.P) { pp = 0; ++n; } }
        }
    }
}

// ---------------------------------------------------------------------------
// Pooling / FC / LSTM (NHWC)
// ---------------------------------------------------------------------------
__global__ void maxpool_kernel(const __bf16* __restrict__ X, __bf16* __restrict__ Y,
                               int N, int C, int H, int W, int P, int Q) {
    int idx = blockIdx.x * blockDim.x + threadIdx.x;
    int total = N * P * Q * C;
    if (idx >= total) return;
    int c = idx % C; int t = idx / C;
    int q = t % Q;   t /= Q;
    int pp = t % P;  int n = t / P;
    float best = -INFINITY;
    for (int r = 0; r < 3; ++r) {
        int hin = pp * 2 - 1 + r;
        if (hin < 0 || hin >= H) continue;
        for (int s = 0; s < 3; ++s) {
            int win = q * 2 - 1 + s;
            if (win < 0 || win >= W) continue;
            best = fmaxf(best, bf2f(X[((size_t)(n * H + hin) * W + win) * C + c]));
        }
    }
    Y[idx] = f2bf(best);
}

// X: [N][HW][C] -> Y: [N][C] (f32 mean over HW)
__global__ void avgpool_kernel(const __bf16* __restrict__ X, float* __restrict__ Y,
                               int N, int C, int HW) {
    int i = blockIdx.x * blockDim.x + threadIdx.x;
    if (i >= N * C) return;
    int n = i / C, c = i % C;
    float s = 0.f;
    for (int k = 0; k < HW; ++k) s += bf2f(X[((size_t)n * HW + k) * C + c]);
    Y[i] = s / (float)HW;
}

// Y[row*y_stride + y_off + o] = dot(X[row,:], W[o,:]) + bias[o]
__global__ void fc_rows_kernel(const float* __restrict__ X, const float* __restrict__ Wm,
                               const float* __restrict__ bias, float* __restrict__ Y,
                               int rows, int in_dim, int out_dim, int y_stride, int y_off) {
    int idx = blockIdx.x * blockDim.x + threadIdx.x;
    if (idx >= rows * out_dim) return;
    int o = idx % out_dim, r = idx / out_dim;
    const float* xr = X + (size_t)r * in_dim;
    const float* wr = Wm + (size_t)o * in_dim;
    float s = bias[o];
    for (int k = 0; k < in_dim; ++k) s += xr[k] * wr[k];
    Y[(size_t)r * y_stride + y_off + o] = s;
}

__device__ __forceinline__ float sigmoidf(float x) { return 1.f / (1.f + __expf(-x)); }

// one block of 512 threads: (b = tid>>7, h = tid&127); feat [4][8][192]
__global__ void lstm_kernel(const float* __restrict__ feat,
                            const float* __restrict__ Wih, const float* __restrict__ Whh,
                            const float* __restrict__ bih, const float* __restrict__ bhh,
                            float* __restrict__ h_out) {
    __shared__ float hs[4][128];
    const int tid = threadIdx.x;
    const int b = tid >> 7, h = tid & 127;
    float c = 0.f;
    hs[b][h] = 0.f;
    __syncthreads();
    for (int t = 0; t < 8; ++t) {
        const float* xt = feat + (size_t)(b * 8 + t) * 192;
        float zi = bih[h]       + bhh[h];
        float zf = bih[128 + h] + bhh[128 + h];
        float zg = bih[256 + h] + bhh[256 + h];
        float zo = bih[384 + h] + bhh[384 + h];
        for (int k = 0; k < 192; ++k) {
            float xv = xt[k];
            zi += Wih[(size_t)(h)       * 192 + k] * xv;
            zf += Wih[(size_t)(128 + h) * 192 + k] * xv;
            zg += Wih[(size_t)(256 + h) * 192 + k] * xv;
            zo += Wih[(size_t)(384 + h) * 192 + k] * xv;
        }
        for (int k = 0; k < 128; ++k) {
            float hv = hs[b][k];
            zi += Whh[(size_t)(h)       * 128 + k] * hv;
            zf += Whh[(size_t)(128 + h) * 128 + k] * hv;
            zg += Whh[(size_t)(256 + h) * 128 + k] * hv;
            zo += Whh[(size_t)(384 + h) * 128 + k] * hv;
        }
        c = sigmoidf(zf) * c + sigmoidf(zi) * tanhf(zg);
        float hn = sigmoidf(zo) * tanhf(c);
        __syncthreads();
        hs[b][h] = hn;
        __syncthreads();
    }
    h_out[b * 128 + h] = hs[b][h];
}

// ---------------------------------------------------------------------------
// Host-side orchestration
// ---------------------------------------------------------------------------
struct BNp    { const float *b, *g, *m, *v; };
struct RBlockP{ BNp bn1, bn2; const float *conv1, *conv2, *dconv; BNp dbn; bool down; };
struct RNetP  { BNp bn1; const float *conv1, *fc_b, *fc_w; RBlockP blk[4][2]; };

static void launch_conv(hipStream_t s, const __bf16* X, const __bf16* Bp, BNp bn,
                        const __bf16* res, __bf16* Y,
                        int N, int C, int H, int W, int K, int R, int Sd,
                        int stride, int pad, int relu) {
    ConvParams cp;
    cp.X = X; cp.Bp = Bp;
    cp.bn_g = bn.g; cp.bn_b = bn.b; cp.bn_m = bn.m; cp.bn_v = bn.v;
    cp.res = res; cp.Y = Y;
    cp.N = N; cp.C = C; cp.H = H; cp.W = W; cp.K = K; cp.R = R; cp.S = Sd;
    cp.stride = stride; cp.pad = pad;
    cp.P = (H + 2 * pad - R) / stride + 1;
    cp.Q = (W + 2 * pad - Sd) / stride + 1;
    cp.relu = relu;
    cp.nKS = (C * R * Sd + 31) / 32;
    int M = N * cp.P * cp.Q;
    dim3 grid((M + 127) / 128, (K + 63) / 64);
    conv_wmma_kernel<<<grid, 128, 0, s>>>(cp);
}

static void run_resnet(hipStream_t stream, const RNetP& net, int out_dim,
                       const __bf16* input,           // [32,224,224,3] bf16 NHWC
                       __bf16* wbuf,                  // packed-weight scratch
                       __bf16* bufA, __bf16* bufB, __bf16* bufC, __bf16* bufD,
                       float* pooled, float* feat, int feat_stride, int feat_off) {
    __bf16* wcur = wbuf;
    auto packB = [&](const float* w, int K, int C, int R, int S) -> const __bf16* {
        int nKS = (C * R * S + 31) / 32;
        __bf16* dst = wcur;
        int total = (K / 16) * nKS * 32;
        pack_b_kernel<<<(total + 255) / 256, 256, 0, stream>>>(w, dst, K, C, R, S, nKS);
        wcur += (size_t)K * nKS * 32;
        return dst;
    };
    const int NI = 32;

    // conv1 7x7/2 + BN + ReLU -> [32,112,112,64]
    const __bf16* w1 = packB(net.conv1, 64, 3, 7, 7);
    launch_conv(stream, input, w1, net.bn1, nullptr, bufA, NI, 3, 224, 224, 64, 7, 7, 2, 3, 1);

    // maxpool 3x3/2 pad1 -> [32,56,56,64]
    {
        int total = NI * 56 * 56 * 64;
        maxpool_kernel<<<(total + 255) / 256, 256, 0, stream>>>(bufA, bufB, NI, 64, 112, 112, 56, 56);
    }

    __bf16 *cur = bufB, *t1 = bufA, *t2 = bufC, *tD = bufD;
    int in_c = 64, hw = 56;
    const int chs[4]     = {64, 128, 256, 512};
    const int strides[4] = {1, 2, 2, 2};
    for (int si = 0; si < 4; ++si) {
        for (int bi = 0; bi < 2; ++bi) {
            const RBlockP& b = net.blk[si][bi];
            int c  = chs[si];
            int st = (bi == 0) ? strides[si] : 1;
            int oh = hw / st;
            const __bf16* res;
            if (b.down) {
                const __bf16* wd = packB(b.dconv, c, in_c, 1, 1);
                launch_conv(stream, cur, wd, b.dbn, nullptr, tD, NI, in_c, hw, hw, c, 1, 1, st, 0, 0);
                res = tD;
            } else {
                res = cur;
            }
            const __bf16* wc1 = packB(b.conv1, c, in_c, 3, 3);
            launch_conv(stream, cur, wc1, b.bn1, nullptr, t1, NI, in_c, hw, hw, c, 3, 3, st, 1, 1);
            const __bf16* wc2 = packB(b.conv2, c, c, 3, 3);
            launch_conv(stream, t1, wc2, b.bn2, res, t2, NI, c, oh, oh, c, 3, 3, 1, 1, 1);
            __bf16* old = cur; cur = t2; t2 = old;   // rotate
            in_c = c; hw = oh;
        }
    }

    // global avg pool [32,7,7,512] -> [32,512], then fc -> feat slice
    avgpool_kernel<<<(NI * 512 + 255) / 256, 256, 0, stream>>>(cur, pooled, NI, 512, 49);
    int tot = NI * out_dim;
    fc_rows_kernel<<<(tot + 255) / 256, 256, 0, stream>>>(pooled, net.fc_w, net.fc_b, feat,
                                                          NI, 512, out_dim, feat_stride, feat_off);
}

extern "C" void kernel_launch(void* const* d_in, const int* in_sizes, int n_in,
                              void* d_out, int out_size, void* d_ws, size_t ws_size,
                              hipStream_t stream) {
    (void)in_sizes; (void)n_in; (void)out_size; (void)ws_size;
    const int B = 4, V = 8;
    const int HWpix = 224 * 224;
    const int CHW = 3 * HWpix;

    // ---- parse inputs (jax tree_flatten: dict keys sorted at every level) ----
    int ii = 0;
    const float* x = (const float*)d_in[ii++];
    auto getBN = [&]() -> BNp {
        BNp r;
        r.b = (const float*)d_in[ii++];
        r.g = (const float*)d_in[ii++];
        r.m = (const float*)d_in[ii++];
        r.v = (const float*)d_in[ii++];
        return r;
    };
    auto getNet = [&]() -> RNetP {
        RNetP n;
        n.bn1   = getBN();                      // "bn1"
        n.conv1 = (const float*)d_in[ii++];     // "conv1"
        n.fc_b  = (const float*)d_in[ii++];     // "fc_b"
        n.fc_w  = (const float*)d_in[ii++];     // "fc_w"
        for (int s = 0; s < 4; ++s)             // "stages"
            for (int bk = 0; bk < 2; ++bk) {
                RBlockP& blk = n.blk[s][bk];
                blk.down  = (bk == 0 && s > 0);
                blk.bn1   = getBN();
                blk.bn2   = getBN();
                blk.conv1 = (const float*)d_in[ii++];
                blk.conv2 = (const float*)d_in[ii++];
                blk.dconv = nullptr;
                if (blk.down) {
                    blk.dbn   = getBN();
                    blk.dconv = (const float*)d_in[ii++];
                }
            }
        return n;
    };
    RNetP net1 = getNet();                        // params1 (fc 128)
    RNetP net2 = getNet();                        // params2 (fc 64)
    const float* Whh  = (const float*)d_in[ii++]; // lstm: Whh < Wih < bhh < bih
    const float* Wih  = (const float*)d_in[ii++];
    const float* bhh  = (const float*)d_in[ii++];
    const float* bih  = (const float*)d_in[ii++];
    const float* fc1w = (const float*)d_in[ii++];
    const float* fc1b = (const float*)d_in[ii++];

    // ---- workspace carve (bump allocator, 256B aligned) ----
    char* ws = (char*)d_ws;
    size_t off = 0;
    auto carve = [&](size_t bytes) -> void* {
        off = (off + 255) & ~(size_t)255;
        void* p = ws + off;
        off += bytes;
        return p;
    };
    float*  d_mean = (float*)carve(32 * sizeof(float));
    int*    d_perm = (int*)carve(32 * sizeof(int));
    size_t  imgEl  = (size_t)B * V * CHW;
    __bf16* xs   = (__bf16*)carve(imgEl * 2);      // NHWC
    __bf16* xr   = (__bf16*)carve(imgEl * 2);      // NHWC
    size_t  actEl = (size_t)32 * 112 * 112 * 64;
    __bf16* bufA = (__bf16*)carve(actEl * 2);
    __bf16* bufB = (__bf16*)carve(actEl * 2);
    __bf16* bufC = (__bf16*)carve(actEl * 2);
    __bf16* bufD = (__bf16*)carve(actEl * 2);
    __bf16* wbuf = (__bf16*)carve((size_t)13 * 1024 * 1024 * 2);
    float*  pooled = (float*)carve((size_t)32 * 512 * 4);
    float*  feat   = (float*)carve((size_t)4 * 8 * 192 * 4);
    float*  hout   = (float*)carve((size_t)4 * 128 * 4);

    // ---- preprocessing: means -> inverse perm -> sorted views + absdiff ----
    view_mean_kernel<<<32, 256, 0, stream>>>(x, d_mean, CHW);
    rank_perm_kernel<<<1, 32, 0, stream>>>(d_mean, d_perm, B, V);
    int totPix = B * HWpix;
    sort_avg_diff_kernel<<<(totPix + 255) / 256, 256, 0, stream>>>(
        x, d_perm, xs, xr, B, V, 3, HWpix);

    // ---- two ResNet-18 streams (WMMA implicit GEMM), feat = concat(f1,f2) ----
    run_resnet(stream, net1, 128, xs, wbuf, bufA, bufB, bufC, bufD, pooled, feat, 192, 0);
    run_resnet(stream, net2,  64, xr, wbuf, bufA, bufB, bufC, bufD, pooled, feat, 192, 128);

    // ---- LSTM over the 8 views, then classifier ----
    lstm_kernel<<<1, 512, 0, stream>>>(feat, Wih, Whh, bih, bhh, hout);
    fc_rows_kernel<<<(4 * 100 + 255) / 256, 256, 0, stream>>>(
        hout, fc1w, fc1b, (float*)d_out, 4, 128, 100, 100, 0);
}
